// Model2BD_89584427860248
// MI455X (gfx1250) — matrix operand order repaired and checked by reading, not compiled, not run on hardware
//
#include <hip/hip_runtime.h>
#include <math.h>

// ---------------- problem constants ----------------
#define NCH   128     // hidden width
#define NDT   45      // distance towers
#define NAT   7       // angle towers
#define NTOW  (NDT + NAT)
#define BFRM  256     // frames per block (8 waves x 32 frames)
#define SH    136     // padded LDS row stride in f16 elems (136*2B = 272B, 16B aligned)
#define REPELCONST 27680.640625f   // 5.5^6

typedef _Float16 v16h __attribute__((ext_vector_type(16)));
typedef _Float16 v8h  __attribute__((ext_vector_type(8)));
typedef float    v8f  __attribute__((ext_vector_type(8)));

// ---- branch-free tanh (native v_tanh_f32 if the builtin exists) ----
__device__ __forceinline__ float fast_tanh(float x) {
#if __has_builtin(__builtin_amdgcn_tanhf)
  return __builtin_amdgcn_tanhf(x);
#else
  const float ax = __builtin_fabsf(x);
  // e = exp(-2*ax) = 2^(-2*log2(e)*ax); underflows to 0 for large ax -> tanh = 1
  const float e = __builtin_amdgcn_exp2f(-2.885390081777927f * ax);
  const float r = __builtin_amdgcn_rcpf(1.0f + e);
  const float t = fmaf(-2.0f * e, r, 1.0f);
  return __builtin_copysignf(t, x);
#endif
}

// ---- fragment loaders (CDNA5 wave32 WMMA 16x16x32 f16 layouts) ----
// A (16x32, MxK): lane L -> row M=L&15; half=L>>4;
//   a[0..7]  = H[M][k0 + 8*half + i]
//   a[8..15] = H[M][k0 + 16 + 8*half + i]
__device__ __forceinline__ v16h load_a(const _Float16* base, int lane, int k0) {
  const int M = lane & 15, h = lane >> 4;
  const _Float16* r = base + M * SH + k0 + 8 * h;
  v8h lo = *(const v8h*)(r);
  v8h hi = *(const v8h*)(r + 16);
  v16h o;
#pragma unroll
  for (int i = 0; i < 8; ++i) { o[i] = lo[i]; o[i + 8] = hi[i]; }
  return o;
}

// B (32x16, KxN): lane L -> col N=L&15; half=L>>4; b[0..7] = W[k0 + 8*half + i][N], b[8..15] = W[k0 + 16 + 8*half + i][N].
// Weights stored transposed in LDS as wl[m*SH + n], so this is a contiguous read.
__device__ __forceinline__ v16h load_b(const _Float16* wl, int lane, int nt, int k0) {
  const int N = lane & 15, h = lane >> 4;
  const _Float16* r = wl + (nt * 16 + N) * SH + k0 + 8 * h;
  v8h lo = *(const v8h*)(r);
  v8h hi = *(const v8h*)(r + 16);
  v16h o;
#pragma unroll
  for (int i = 0; i < 8; ++i) { o[i] = lo[i]; o[i + 8] = hi[i]; }
  return o;
}

// One hidden layer for this wave's 32 frames: h <- tanh(h @ W + b), in-place in LDS.
// hwave: base of this wave's 32 activation rows. wl: transposed weights [m*SH+n].
__device__ __forceinline__ void hidden_layer(_Float16* hwave, const _Float16* wl,
                                             const float* bias, int lane) {
  // Preload all A fragments so the in-place update is race-free within the wave.
  v16h A[2][4];
#pragma unroll
  for (int mt = 0; mt < 2; ++mt)
#pragma unroll
    for (int kb = 0; kb < 4; ++kb)
      A[mt][kb] = load_a(hwave + mt * 16 * SH, lane, kb * 32);

  const int col = lane & 15, half = lane >> 4;
  for (int nt = 0; nt < 8; ++nt) {
    v16h Bf[4];
#pragma unroll
    for (int kb = 0; kb < 4; ++kb) Bf[kb] = load_b(wl, lane, nt, kb * 32);
    const float bb = bias[nt * 16 + col];
#pragma unroll
    for (int mt = 0; mt < 2; ++mt) {
      v8f acc = {};
#pragma unroll
      for (int kb = 0; kb < 4; ++kb)
        acc = __builtin_amdgcn_wmma_f32_16x16x32_f16(
            false, A[mt][kb], false, Bf[kb], (short)0, acc, false, false);
      // D layout: lane col = nt*16+(L&15); VGPR v -> row v + 8*half
      _Float16* dr = hwave + (mt * 16 + 8 * half) * SH + nt * 16 + col;
#pragma unroll
      for (int v = 0; v < 8; ++v)
        dr[v * SH] = (_Float16)fast_tanh(acc[v] + bb);
    }
  }
}

__global__ __launch_bounds__(256) void tower_kernel(
    const float* __restrict__ F_dist, const float* __restrict__ F_cos,
    const float* __restrict__ F_sin,
    const float* __restrict__ Zd, const float* __restrict__ Zc,
    const float* __restrict__ Zs,
    const float* __restrict__ W1_d, const float* __restrict__ b1_d,
    const float* __restrict__ Wh_d, const float* __restrict__ bh_d,
    const float* __restrict__ Wo_d, const float* __restrict__ bo_d,
    const float* __restrict__ W1_a, const float* __restrict__ b1_a,
    const float* __restrict__ Wh_a, const float* __restrict__ bh_a,
    const float* __restrict__ Wo_a, const float* __restrict__ bo_a,
    float* __restrict__ out) {
  __shared__ __align__(16) _Float16 hbuf[BFRM * SH];        // activations, f16
  __shared__ __align__(16) _Float16 wbuf[2][NCH * SH];      // Wh^T per layer, f16
  __shared__ float w1s[2][NCH], b1s[NCH], bhs[2][NCH], wos[NCH];

  const int tid = threadIdx.x;
  const int tower = blockIdx.y;
  const long frame0 = (long)blockIdx.x * BFRM;
  const bool isAngle = (tower >= NDT);
  const int f = isAngle ? tower - NDT : tower;
  const int ntw = isAngle ? NAT : NDT;

  const float* W1 = isAngle ? (W1_a + (size_t)f * 2 * NCH) : (W1_d + (size_t)f * NCH);
  const float* b1 = isAngle ? (b1_a + (size_t)f * NCH) : (b1_d + (size_t)f * NCH);
  const float* Wh = isAngle ? (Wh_a + (size_t)f * NCH * NCH) : (Wh_d + (size_t)f * NCH * NCH);
  const float* bh = isAngle ? (bh_a + (size_t)f * NCH) : (bh_d + (size_t)f * NCH);
  const float* Wo = isAngle ? (Wo_a + (size_t)f * NCH) : (Wo_d + (size_t)f * NCH);
  const float bo = isAngle ? bo_a[f] : bo_d[f];

  // ---- cooperative weight staging ----
  for (int i = tid; i < NCH; i += 256) {
    w1s[0][i] = W1[i];
    w1s[1][i] = isAngle ? W1[NCH + i] : 0.0f;
    b1s[i] = b1[i];
    bhs[0][i] = bh[i];
    bhs[1][i] = bh[(size_t)ntw * NCH + i];
    wos[i] = Wo[i];
  }
#pragma unroll
  for (int l = 0; l < 2; ++l) {
    const float* g = Wh + (size_t)l * ntw * NCH * NCH;   // [n][m] row-major
    for (int i = tid; i < NCH * NCH; i += 256) {
      const int n = i >> 7, m = i & 127;
      wbuf[l][m * SH + n] = (_Float16)g[i];              // store transposed [m][n]
    }
  }
  __syncthreads();

  // ---- layer 1: rank-1/rank-2 input layer, one thread per frame ----
  {
    const long frame = frame0 + tid;
    float x0, x1 = 0.0f;
    if (!isAngle) {
      x0 = (F_dist[frame * NDT + f] - Zd[f]) / Zd[NDT + f];
    } else {
      x0 = (F_cos[frame * NAT + 0] - Zc[0]) / Zc[NAT];   // INDEX5 == 0: column 0
      x1 = (F_sin[frame * NAT + 0] - Zs[0]) / Zs[NAT];
    }
    _Float16* hr = hbuf + tid * SH;
    for (int n = 0; n < NCH; n += 2) {
      const float v0 = fmaf(x0, w1s[0][n],     fmaf(x1, w1s[1][n],     b1s[n]));
      const float v1 = fmaf(x0, w1s[0][n + 1], fmaf(x1, w1s[1][n + 1], b1s[n + 1]));
      union { _Float16 h[2]; unsigned u; } p;
      p.h[0] = (_Float16)fast_tanh(v0);
      p.h[1] = (_Float16)fast_tanh(v1);
      *(unsigned*)(hr + n) = p.u;     // b32 store (4B aligned: n even, row 272B)
    }
  }
  __syncthreads();

  // ---- hidden layers via WMMA: each wave owns 32 frames ----
  const int wave = tid >> 5, lane = tid & 31;
  _Float16* hwave = hbuf + wave * 32 * SH;
  hidden_layer(hwave, wbuf[0], bhs[0], lane);
  __syncthreads();
  hidden_layer(hwave, wbuf[1], bhs[1], lane);
  __syncthreads();

  // ---- output layer + reduction ----
  float val = bo;
  {
    const v8h* hv = (const v8h*)(hbuf + tid * SH);
    for (int c = 0; c < NCH / 8; ++c) {
      const v8h h = hv[c];
#pragma unroll
      for (int i = 0; i < 8; ++i) val = fmaf((float)h[i], wos[c * 8 + i], val);
    }
  }
#pragma unroll
  for (int off = 16; off > 0; off >>= 1) val += __shfl_down(val, off, 32);
  if (lane == 0) atomicAdd(out, val);
}

__global__ __launch_bounds__(256) void bond_repel_kernel(
    const float* __restrict__ F_dist, const float* __restrict__ F_angle,
    const float* __restrict__ BondConst, float* __restrict__ out, int B) {
  const long b = (long)blockIdx.x * 256 + threadIdx.x;
  float val = 0.0f;
  if (b < B) {
    const float* d = F_dist + b * NDT;
    const float* a = F_angle + b * NAT;
    float bond = 0.0f;
#pragma unroll
    for (int j = 0; j < 9; ++j) {
      const float u = d[j] - BondConst[16 + j];
      bond = fmaf(BondConst[j] * u, u, bond);
    }
#pragma unroll
    for (int j = 0; j < 7; ++j) {
      const float u = a[j] - BondConst[16 + 9 + j];
      bond = fmaf(BondConst[9 + j] * u, u, bond);
    }
    val = 0.5f * bond;
    for (int j = 17; j < 45; ++j) {
      const float x = d[j];
      const float x2 = x * x;
      const float x6 = x2 * x2 * x2;
      val += REPELCONST / x6;
    }
  }
#pragma unroll
  for (int off = 16; off > 0; off >>= 1) val += __shfl_down(val, off, 32);
  if ((threadIdx.x & 31) == 0) atomicAdd(out, val);
}

__global__ void zero_kernel(float* out) { out[0] = 0.0f; }

extern "C" void kernel_launch(void* const* d_in, const int* in_sizes, int n_in,
                              void* d_out, int out_size, void* d_ws, size_t ws_size,
                              hipStream_t stream) {
  const float* F_dist  = (const float*)d_in[0];
  const float* F_cos   = (const float*)d_in[1];
  const float* F_sin   = (const float*)d_in[2];
  const float* F_angle = (const float*)d_in[3];
  const float* Zd      = (const float*)d_in[4];
  const float* Zc      = (const float*)d_in[5];
  const float* Zs      = (const float*)d_in[6];
  const float* BondC   = (const float*)d_in[7];
  const float* W1_d    = (const float*)d_in[8];
  const float* b1_d    = (const float*)d_in[9];
  const float* Wh_d    = (const float*)d_in[10];
  const float* bh_d    = (const float*)d_in[11];
  const float* Wo_d    = (const float*)d_in[12];
  const float* bo_d    = (const float*)d_in[13];
  const float* W1_a    = (const float*)d_in[14];
  const float* b1_a    = (const float*)d_in[15];
  const float* Wh_a    = (const float*)d_in[16];
  const float* bh_a    = (const float*)d_in[17];
  const float* Wo_a    = (const float*)d_in[18];
  const float* bo_a    = (const float*)d_in[19];
  float* out = (float*)d_out;

  const int B = in_sizes[0] / NDT;           // 262144
  const int fblocks = B / BFRM;              // 1024

  zero_kernel<<<1, 1, 0, stream>>>(out);
  bond_repel_kernel<<<B / 256, 256, 0, stream>>>(F_dist, F_angle, BondC, out, B);
  dim3 grid(fblocks, NTOW);
  tower_kernel<<<grid, 256, 0, stream>>>(
      F_dist, F_cos, F_sin, Zd, Zc, Zs,
      W1_d, b1_d, Wh_d, bh_d, Wo_d, bo_d,
      W1_a, b1_a, Wh_a, bh_a, Wo_a, bo_a, out);
}
